// OtLoss_7086696038716
// MI455X (gfx1250) — matrix operand order repaired and checked by reading, not compiled, not run on hardware
//
#include <hip/hip_runtime.h>

#define NN 4096
#define MM 4096
#define DD 2048
#define ALPHA 0.25f

typedef __attribute__((ext_vector_type(16))) __bf16 v16bf;
typedef __attribute__((ext_vector_type(8)))  float  v8f;

__device__ __forceinline__ unsigned short f2bf(float f) {
  unsigned int u = __float_as_uint(f);
  u += 0x7FFFu + ((u >> 16) & 1u);   // round-to-nearest-even
  return (unsigned short)(u >> 16);
}

__device__ __forceinline__ unsigned pack2bf(float a, float b) {
  return (unsigned)f2bf(a) | ((unsigned)f2bf(b) << 16);
}

__device__ __forceinline__ float waveRed(float v) {
#pragma unroll
  for (int o = 16; o > 0; o >>= 1) v += __shfl_xor(v, o, 32);
  return v;
}

// ---- CDNA5 async copy: global -> LDS, 16B per lane, tracked by ASYNCcnt ----
__device__ __forceinline__ void async_ld_b128(unsigned ldsOff, const void* g) {
  asm volatile("global_load_async_to_lds_b128 %0, %1, off"
               :: "v"(ldsOff), "v"(g) : "memory");
}
__device__ __forceinline__ void wait_async6() {
  asm volatile("s_wait_asynccnt 0x6" ::: "memory");
}
__device__ __forceinline__ void wait_async0() {
  asm volatile("s_wait_asynccnt 0x0" ::: "memory");
}

__global__ void k_zero(float* __restrict__ out) {
  if (threadIdx.x == 0) out[0] = 0.f;
}

// gamma fp32 [N,M] -> bf16 [N,M] workspace
__global__ void k_cvt_gamma(const float* __restrict__ G, unsigned short* __restrict__ gBF) {
  const size_t i4 = ((size_t)blockIdx.x * 256 + threadIdx.x) * 4;
  const float4 v = *(const float4*)(G + i4);
  uint2 p;
  p.x = pack2bf(v.x, v.y);
  p.y = pack2bf(v.z, v.w);
  *(uint2*)(gBF + i4) = p;
}

// t fp32 [M,D] -> bf16 transposed [D,M] workspace (32x32 LDS tile transpose)
__global__ void k_cvt_tT(const float* __restrict__ T, unsigned short* __restrict__ tT) {
  __shared__ unsigned short tl[32][33];
  const int tid = threadIdx.x;
  const int mbase = blockIdx.x * 32;
  const int dbase = blockIdx.y * 32;
  {
    const int row = tid >> 3;          // m-local
    const int c4  = (tid & 7) * 4;     // d-local
    const float4 v = *(const float4*)(T + (size_t)(mbase + row) * DD + dbase + c4);
    tl[row][c4 + 0] = f2bf(v.x); tl[row][c4 + 1] = f2bf(v.y);
    tl[row][c4 + 2] = f2bf(v.z); tl[row][c4 + 3] = f2bf(v.w);
  }
  __syncthreads();
  {
    const int drow = tid >> 3;         // d-local
    const int c4   = (tid & 7) * 4;    // m-local
    uint2 p;
    p.x = (unsigned)tl[c4 + 0][drow] | ((unsigned)tl[c4 + 1][drow] << 16);
    p.y = (unsigned)tl[c4 + 2][drow] | ((unsigned)tl[c4 + 3][drow] << 16);
    *(uint2*)(tT + (size_t)(dbase + drow) * MM + mbase + c4) = p;
  }
}

// ---------------------------------------------------------------------------
// Fast cross term: P = gamma(N,M) @ t(M,D); adds -2*ALPHA*sum(P .* s).
// bf16 operands pre-staged in workspace; main loop = double-buffered async
// global->LDS B128 copies overlapped with 16 WMMAs per wave (64x64 tile).
// Block tile: 256 (n) x 128 (d), 8 waves.
// ---------------------------------------------------------------------------
#define APITCH 40   // ushorts per A row (32 data + pad): 80B, 16B aligned, conflict-free
#define BPITCH 40
#define ABUFB  (256 * APITCH * 2)  // bytes per A buffer
#define BBUFB  (128 * BPITCH * 2)  // bytes per B buffer

__launch_bounds__(256)
__global__ void k_cross_async(const float* __restrict__ S,
                              const unsigned short* __restrict__ gBF,
                              const unsigned short* __restrict__ tT,
                              float* __restrict__ out) {
  __shared__ unsigned short sA[2][256 * APITCH];  // gamma tile [row n][k m]
  __shared__ unsigned short sB[2][128 * BPITCH];  // tT tile   [col d][k m]
  __shared__ float sred[8];

  const int tid   = threadIdx.x;
  const int nbase = blockIdx.y * 256;
  const int dbase = blockIdx.x * 128;

  const int w    = tid >> 5;
  const int wr   = w & 3;        // 4 row groups x 64 rows
  const int wc   = w >> 2;       // 2 col groups x 64 cols
  const int lane = tid & 31;
  const int l15  = lane & 15;
  const int hf   = lane >> 4;

  const unsigned baseA = (unsigned)(size_t)(void*)&sA[0][0];
  const unsigned baseB = (unsigned)(size_t)(void*)&sB[0][0];

  v8f acc[4][4];
  const v8f vzero = {0.f, 0.f, 0.f, 0.f, 0.f, 0.f, 0.f, 0.f};
#pragma unroll
  for (int i = 0; i < 4; ++i)
#pragma unroll
    for (int j = 0; j < 4; ++j) acc[i][j] = vzero;

  // issue one k-step's tiles (6 async B128 per thread: 4 for A, 2 for B)
  auto issue = [&](int buf, int kk) {
    const int mbase = kk * 32;
#pragma unroll
    for (int p = 0; p < 4; ++p) {            // A: 256 rows x 4 chunks
      const int chunk = tid + p * 256;
      const int row = chunk >> 2, sub = chunk & 3;
      const unsigned short* g = gBF + (size_t)(nbase + row) * MM + mbase + sub * 8;
      async_ld_b128(baseA + buf * ABUFB + row * (APITCH * 2) + sub * 16, g);
    }
#pragma unroll
    for (int p = 0; p < 2; ++p) {            // B: 128 rows x 4 chunks
      const int chunk = tid + p * 256;
      const int row = chunk >> 2, sub = chunk & 3;
      const unsigned short* g = tT + (size_t)(dbase + row) * MM + mbase + sub * 8;
      async_ld_b128(baseB + buf * BBUFB + row * (BPITCH * 2) + sub * 16, g);
    }
  };

  issue(0, 0);
  const int KSTEPS = MM / 32;
  for (int kk = 0; kk < KSTEPS; ++kk) {
    const int b = kk & 1;
    if (kk + 1 < KSTEPS) { issue(b ^ 1, kk + 1); wait_async6(); }
    else                 { wait_async0(); }
    __syncthreads();

    union FB { unsigned int u[8]; v16bf v; };
    FB fb[4];
#pragma unroll
    for (int ct = 0; ct < 4; ++ct) {
      const int col = wc * 64 + ct * 16 + l15;
#pragma unroll
      for (int j = 0; j < 8; ++j) {
        const int k0 = 2 * j + (j >= 4 ? 8 : 0) + 8 * hf;     // B: same K layout as A (+8 hi lanes)
        fb[ct].u[j] = *(const unsigned int*)&sB[b][col * BPITCH + k0];
      }
    }
#pragma unroll
    for (int rt = 0; rt < 4; ++rt) {
      FB fa;
      const int row = wr * 64 + rt * 16 + l15;
#pragma unroll
      for (int j = 0; j < 8; ++j) {
        // ISA A layout: vgpr j holds K = 2j (j<4) / 16+2(j-4) (j>=4), +8 hi lanes
        const int k0 = 2 * j + (j >= 4 ? 8 : 0) + 8 * hf;
        fa.u[j] = *(const unsigned int*)&sA[b][row * APITCH + k0];
      }
#pragma unroll
      for (int ct = 0; ct < 4; ++ct) {
        acc[rt][ct] = __builtin_amdgcn_wmma_f32_16x16x32_bf16(
            false, fa.v, false, fb[ct].v, (short)0, acc[rt][ct], false, false);
      }
    }
    __syncthreads();
  }

  // epilogue: dot with s tile, reduce, atomic add
  float part = 0.f;
#pragma unroll
  for (int rt = 0; rt < 4; ++rt) {
#pragma unroll
    for (int ct = 0; ct < 4; ++ct) {
      const int d = dbase + wc * 64 + ct * 16 + l15;
#pragma unroll
      for (int j = 0; j < 8; ++j) {
        const int n = nbase + wr * 64 + rt * 16 + j + 8 * hf;  // C/D f32 layout
        part += acc[rt][ct][j] * S[(size_t)n * DD + d];
      }
    }
  }
  part = waveRed(part);
  if (lane == 0) sred[w] = part;
  __syncthreads();
  if (tid == 0) {
    float tot = 0.f;
#pragma unroll
    for (int i = 0; i < 8; ++i) tot += sred[i];
    atomicAdd(out, -2.0f * ALPHA * tot);
  }
}

// ---------------------------------------------------------------------------
// Fallback cross kernel (no workspace): fp32->bf16 conversion fused in-loop.
// ---------------------------------------------------------------------------
__launch_bounds__(256)
__global__ void k_cross_fb(const float* __restrict__ S, const float* __restrict__ T,
                           const float* __restrict__ G, float* __restrict__ out) {
  __shared__ unsigned short sA[128 * 34];
  __shared__ unsigned short sB[128 * 34];
  __shared__ float sred[8];

  const int tid   = threadIdx.x;
  const int nbase = blockIdx.y * 128;
  const int dbase = blockIdx.x * 128;
  const int w = tid >> 5, wr = w & 1, wc = w >> 1;
  const int lane = tid & 31, l15 = lane & 15, hf = lane >> 4;

  v8f acc[4][2];
  const v8f vzero = {0.f, 0.f, 0.f, 0.f, 0.f, 0.f, 0.f, 0.f};
#pragma unroll
  for (int i = 0; i < 4; ++i)
#pragma unroll
    for (int j = 0; j < 2; ++j) acc[i][j] = vzero;

  for (int kk = 0; kk < MM / 32; ++kk) {
    const int mbase = kk * 32;
    __syncthreads();
    {
      const int row0 = tid >> 3, c4 = (tid & 7) * 4;
#pragma unroll
      for (int p = 0; p < 4; ++p) {
        const int row = p * 32 + row0;
        const float4 v = *(const float4*)(G + (size_t)(nbase + row) * MM + mbase + c4);
        unsigned short* dst = &sA[row * 34 + c4];
        dst[0] = f2bf(v.x); dst[1] = f2bf(v.y); dst[2] = f2bf(v.z); dst[3] = f2bf(v.w);
      }
    }
    {
      const int row0 = tid >> 5, c4 = (tid & 31) * 4;
#pragma unroll
      for (int p = 0; p < 4; ++p) {
        const int row = p * 8 + row0;
        const float4 v = *(const float4*)(T + (size_t)(mbase + row) * DD + dbase + c4);
        sB[(c4 + 0) * 34 + row] = f2bf(v.x);
        sB[(c4 + 1) * 34 + row] = f2bf(v.y);
        sB[(c4 + 2) * 34 + row] = f2bf(v.z);
        sB[(c4 + 3) * 34 + row] = f2bf(v.w);
      }
    }
    __syncthreads();

    union FB { unsigned int u[8]; v16bf v; };
    FB fb[2];
#pragma unroll
    for (int ct = 0; ct < 2; ++ct) {
      const int col = wc * 32 + ct * 16 + l15;
#pragma unroll
      for (int j = 0; j < 8; ++j)
        fb[ct].u[j] = *(const unsigned int*)&sB[col * 34 + 2 * j + (j >= 4 ? 8 : 0) + 8 * hf];
    }
#pragma unroll
    for (int rt = 0; rt < 4; ++rt) {
      FB fa;
      const int row = wr * 64 + rt * 16 + l15;
#pragma unroll
      for (int j = 0; j < 8; ++j) {
        const int k0 = 2 * j + (j >= 4 ? 8 : 0) + 8 * hf;
        fa.u[j] = *(const unsigned int*)&sA[row * 34 + k0];
      }
#pragma unroll
      for (int ct = 0; ct < 2; ++ct)
        acc[rt][ct] = __builtin_amdgcn_wmma_f32_16x16x32_bf16(
            false, fa.v, false, fb[ct].v, (short)0, acc[rt][ct], false, false);
    }
  }

  float part = 0.f;
#pragma unroll
  for (int rt = 0; rt < 4; ++rt)
#pragma unroll
    for (int ct = 0; ct < 2; ++ct) {
      const int d = dbase + wc * 32 + ct * 16 + l15;
#pragma unroll
      for (int j = 0; j < 8; ++j) {
        const int n = nbase + wr * 64 + rt * 16 + j + 8 * hf;
        part += acc[rt][ct][j] * S[(size_t)n * DD + d];
      }
    }
  part = waveRed(part);
  if (lane == 0) sred[w] = part;
  __syncthreads();
  if (tid == 0) {
    float tot = 0.f;
#pragma unroll
    for (int i = 0; i < 8; ++i) tot += sred[i];
    atomicAdd(out, -2.0f * ALPHA * tot);
  }
}

// term1: ALPHA * sum_n ||s_n||^2 * rowsum(gamma)_n
__global__ void k_term1(const float* __restrict__ S, const float* __restrict__ G,
                        float* __restrict__ out) {
  __shared__ float sa[8], sb[8];
  const int n = blockIdx.x, tid = threadIdx.x;
  float ss = 0.f, gr = 0.f;
  for (int d = tid; d < DD; d += 256) { float v = S[(size_t)n * DD + d]; ss += v * v; }
  for (int m = tid; m < MM; m += 256) gr += G[(size_t)n * MM + m];
  ss = waveRed(ss); gr = waveRed(gr);
  if ((tid & 31) == 0) { sa[tid >> 5] = ss; sb[tid >> 5] = gr; }
  __syncthreads();
  if (tid == 0) {
    float a = 0.f, b = 0.f;
#pragma unroll
    for (int i = 0; i < 8; ++i) { a += sa[i]; b += sb[i]; }
    atomicAdd(out, ALPHA * a * b);
  }
}

__global__ void k_tsq(const float* __restrict__ T, float* __restrict__ tsq) {
  __shared__ float sred[8];
  const int m = blockIdx.x, tid = threadIdx.x;
  float tq = 0.f;
  for (int d = tid; d < DD; d += 256) { float v = T[(size_t)m * DD + d]; tq += v * v; }
  tq = waveRed(tq);
  if ((tid & 31) == 0) sred[tid >> 5] = tq;
  __syncthreads();
  if (tid == 0) {
    float a = 0.f;
#pragma unroll
    for (int i = 0; i < 8; ++i) a += sred[i];
    tsq[m] = a;
  }
}

__global__ void k_term2(const float* __restrict__ G, const float* __restrict__ tsq,
                        float* __restrict__ out) {
  __shared__ float sred[8];
  const int m  = blockIdx.x * 256 + threadIdx.x;
  const int n0 = blockIdx.y * 128;
  const float tq = tsq[m];
  float a = 0.f;
  for (int r = 0; r < 128; ++r) a += G[(size_t)(n0 + r) * MM + m];
  a *= tq;
  a = waveRed(a);
  if ((threadIdx.x & 31) == 0) sred[threadIdx.x >> 5] = a;
  __syncthreads();
  if (threadIdx.x == 0) {
    float s = 0.f;
#pragma unroll
    for (int i = 0; i < 8; ++i) s += sred[i];
    atomicAdd(out, ALPHA * s);
  }
}

__global__ void k_term2_fb(const float* __restrict__ G, const float* __restrict__ T,
                           float* __restrict__ out) {
  __shared__ float sred[8];
  const int m  = blockIdx.x * 256 + threadIdx.x;
  const int n0 = blockIdx.y * 128;
  float tq = 0.f;
  for (int d = 0; d < DD; ++d) { float v = T[(size_t)m * DD + d]; tq += v * v; }
  float a = 0.f;
  for (int r = 0; r < 128; ++r) a += G[(size_t)(n0 + r) * MM + m];
  a *= tq;
  a = waveRed(a);
  if ((threadIdx.x & 31) == 0) sred[threadIdx.x >> 5] = a;
  __syncthreads();
  if (threadIdx.x == 0) {
    float s = 0.f;
#pragma unroll
    for (int i = 0; i < 8; ++i) s += sred[i];
    atomicAdd(out, ALPHA * s);
  }
}

extern "C" void kernel_launch(void* const* d_in, const int* in_sizes, int n_in,
                              void* d_out, int out_size, void* d_ws, size_t ws_size,
                              hipStream_t stream) {
  (void)in_sizes; (void)n_in; (void)out_size;
  const float* S = (const float*)d_in[0];   // ft_source [N,D]
  const float* T = (const float*)d_in[1];   // ft_target [M,D]
  const float* G = (const float*)d_in[2];   // gamma     [N,M]
  float* out = (float*)d_out;

  const size_t szGamma = (size_t)NN * MM * sizeof(unsigned short);  // 32MB
  const size_t szTT    = (size_t)DD * MM * sizeof(unsigned short);  // 16MB
  const size_t szTsq   = (size_t)MM * sizeof(float);                // 16KB

  k_zero<<<1, 64, 0, stream>>>(out);

  if (ws_size >= szGamma + szTT + szTsq) {
    unsigned short* gBF = (unsigned short*)d_ws;
    unsigned short* tT  = (unsigned short*)((char*)d_ws + szGamma);
    float* tsq          = (float*)((char*)d_ws + szGamma + szTT);

    k_cvt_gamma<<<(NN * MM) / 1024, 256, 0, stream>>>(G, gBF);
    k_cvt_tT<<<dim3(MM / 32, DD / 32), 256, 0, stream>>>(T, tT);
    k_cross_async<<<dim3(DD / 128, NN / 256), 256, 0, stream>>>(S, gBF, tT, out);
    k_term1<<<NN, 256, 0, stream>>>(S, G, out);
    k_tsq<<<MM, 256, 0, stream>>>(T, tsq);
    k_term2<<<dim3(MM / 256, NN / 128), 256, 0, stream>>>(G, tsq, out);
  } else if (ws_size >= szTsq) {
    k_cross_fb<<<dim3(DD / 128, NN / 128), 256, 0, stream>>>(S, T, G, out);
    k_term1<<<NN, 256, 0, stream>>>(S, G, out);
    float* tsq = (float*)d_ws;
    k_tsq<<<MM, 256, 0, stream>>>(T, tsq);
    k_term2<<<dim3(MM / 256, NN / 128), 256, 0, stream>>>(G, tsq, out);
  } else {
    k_cross_fb<<<dim3(DD / 128, NN / 128), 256, 0, stream>>>(S, T, G, out);
    k_term1<<<NN, 256, 0, stream>>>(S, G, out);
    k_term2_fb<<<dim3(MM / 256, NN / 128), 256, 0, stream>>>(G, T, out);
  }
}
